// DecoderCell_6021544149148
// MI455X (gfx1250) — matrix operand order repaired and checked by reading, not compiled, not run on hardware
//
#include <hip/hip_runtime.h>
#include <hip/hip_bf16.h>
#include <math.h>

// =====================================================================
// LFADS-style DecoderCell for MI455X (gfx1250), bf16 WMMA + f32 accum.
// Each wave32 owns 16 batch rows and runs the fully-fused cell:
//   con GRU (K=384/256) -> co linear -> gen GRU (K=96/512) -> factors.
// Cross-layout staging (C-layout accumulators -> A-layout operands)
// goes through per-wave LDS; no inter-wave synchronization needed.
// =====================================================================

typedef __bf16 bf16;
typedef __attribute__((ext_vector_type(16))) __bf16 v16bf;
typedef __attribute__((ext_vector_type(8)))  __bf16 v8bf;
typedef __attribute__((ext_vector_type(8)))  float  v8f;

#define OUTW 1104     // out/h_0 row width
#define INW  272      // input row width

// ws element offsets (bf16)
#define OFF_CONWIH 0                        // 768 x 384
#define OFF_CONWHH (OFF_CONWIH + 294912)    // 768 x 256
#define OFF_COW    (OFF_CONWHH + 196608)    // 128 x 256
#define OFF_GENWIH (OFF_COW    + 32768)     // 1536 x 96 (K padded 80->96)
#define OFF_GENWHH (OFF_GENWIH + 147456)    // 1536 x 512
#define OFF_FACN   (OFF_GENWHH + 786432)    // 128 x 512 (row-normalized)
#define WS_ELEMS   (OFF_FACN   + 65536)     // 1,523,712 bf16 = ~3.05 MB

__device__ __forceinline__ v8f vzero8() {
  v8f x;
#pragma unroll
  for (int i = 0; i < 8; ++i) x[i] = 0.0f;
  return x;
}

__device__ __forceinline__ v8f wmma_bf16(v16bf a, v16bf b, v8f c) {
  // D = A(16x32 bf16) * B(32x16 bf16) + C(16x16 f32)
  return __builtin_amdgcn_wmma_f32_16x16x32_bf16(false, a, false, b,
                                                 (short)0, c, false, false);
}

// A-fragment (16x32 bf16) from row-major f32 source. rowptr = this lane's
// batch row. Elements 0..7 <-> K = k0 + half*8 + i ; 8..15 <-> +16.
__device__ __forceinline__ v16bf a_frag_f32(const float* rowptr, int k0, int half) {
  const float4* p0 = (const float4*)(rowptr + k0 + half * 8);
  const float4* p1 = (const float4*)(rowptr + k0 + 16 + half * 8);
  float4 x0 = p0[0], x1 = p0[1];
  float4 y0 = p1[0], y1 = p1[1];
  v16bf a;
  a[0] = (__bf16)x0.x; a[1] = (__bf16)x0.y; a[2] = (__bf16)x0.z; a[3] = (__bf16)x0.w;
  a[4] = (__bf16)x1.x; a[5] = (__bf16)x1.y; a[6] = (__bf16)x1.z; a[7] = (__bf16)x1.w;
  a[8]  = (__bf16)y0.x; a[9]  = (__bf16)y0.y; a[10] = (__bf16)y0.z; a[11] = (__bf16)y0.w;
  a[12] = (__bf16)y1.x; a[13] = (__bf16)y1.y; a[14] = (__bf16)y1.z; a[15] = (__bf16)y1.w;
  return a;
}

// A-fragment from bf16 row-major LDS staging (rowptr = lane's row base).
__device__ __forceinline__ v16bf a_frag_lds(const bf16* rowptr, int k0, int half) {
  const v8bf* p0 = (const v8bf*)(rowptr + k0 + half * 8);
  const v8bf* p1 = (const v8bf*)(rowptr + k0 + 16 + half * 8);
  v8bf x = *p0, y = *p1;
  v16bf a;
#pragma unroll
  for (int i = 0; i < 8; ++i) { a[i] = x[i]; a[8 + i] = y[i]; }
  return a;
}

// B-fragment (32x16 bf16) from bf16 weights stored row-major (N, K):
// lane n = lane&15, elements 0..7 <-> K = k0 + half*8 + i ; 8..15 <-> +16.
__device__ __forceinline__ v16bf b_frag(const bf16* w, int ldk, int nrow,
                                        int k0, int half) {
  const bf16* base = w + (size_t)nrow * ldk + k0 + half * 8;
  v8bf x = *(const v8bf*)base, y = *(const v8bf*)(base + 16);
  v16bf b;
#pragma unroll
  for (int i = 0; i < 8; ++i) { b[i] = x[i]; b[8 + i] = y[i]; }
  return b;
}

// ---------------------------------------------------------------------
// Prep kernel 1: f32 -> bf16 weight conversion (gen_w_ih padded K 80->96).
// ---------------------------------------------------------------------
__global__ void cvt_weights_kernel(const float* __restrict__ conWih,
                                   const float* __restrict__ conWhh,
                                   const float* __restrict__ coW,
                                   const float* __restrict__ genWih,
                                   const float* __restrict__ genWhh,
                                   bf16* __restrict__ ws) {
  const int N0 = 294912, N1 = 196608, N2 = 32768, N3 = 147456, N4 = 786432;
  const int total = N0 + N1 + N2 + N3 + N4;
  int i = blockIdx.x * blockDim.x + threadIdx.x;
  if (i >= total) return;
  float v;
  if (i < N0) {
    v = conWih[i];
  } else if (i < N0 + N1) {
    v = conWhh[i - N0];
  } else if (i < N0 + N1 + N2) {
    v = coW[i - N0 - N1];
  } else if (i < N0 + N1 + N2 + N3) {
    int d = i - N0 - N1 - N2;
    int row = d / 96, col = d - row * 96;
    v = (col < 80) ? genWih[row * 80 + col] : 0.0f;
  } else {
    v = genWhh[i - N0 - N1 - N2 - N3];
  }
  ws[i] = (__bf16)v;
}

// ---------------------------------------------------------------------
// Prep kernel 2: row-normalize fac_w (128 x 512), store bf16.
// ---------------------------------------------------------------------
__global__ void fac_norm_kernel(const float* __restrict__ facW,
                                bf16* __restrict__ dst) {
  __shared__ float red[128];
  const int row = blockIdx.x;            // 0..127
  const float* src = facW + row * 512;
  float s = 0.0f;
  for (int c = threadIdx.x; c < 512; c += 128) {
    float x = src[c];
    s += x * x;
  }
  red[threadIdx.x] = s;
  __syncthreads();
  for (int off = 64; off > 0; off >>= 1) {
    if (threadIdx.x < off) red[threadIdx.x] += red[threadIdx.x + off];
    __syncthreads();
  }
  const float inv = 1.0f / fmaxf(sqrtf(red[0]), 1e-12f);
  for (int c = threadIdx.x; c < 512; c += 128)
    dst[row * 512 + c] = (__bf16)(src[c] * inv);
}

// ---------------------------------------------------------------------
// Main fused decoder-cell kernel. 128 threads = 4 waves; each wave owns
// 16 batch rows. Dynamic LDS: per-wave 16x512 bf16 A-staging (abuf) +
// 16x512 f16 gate buffer (zbuf, aliased as bf16 for state staging).
// ---------------------------------------------------------------------
__global__ __launch_bounds__(128) void decoder_cell_kernel(
    const float* __restrict__ input, const float* __restrict__ h0,
    const float* __restrict__ eps,
    const float* __restrict__ genBih, const float* __restrict__ genBhh,
    const float* __restrict__ conBih, const float* __restrict__ conBhh,
    const float* __restrict__ coB,
    const bf16* __restrict__ conWih, const bf16* __restrict__ conWhh,
    const bf16* __restrict__ coW, const bf16* __restrict__ genWih,
    const bf16* __restrict__ genWhh, const bf16* __restrict__ facWn,
    float* __restrict__ out) {
  extern __shared__ char smem[];
  const int tid  = threadIdx.x;
  const int wid  = tid >> 5;
  const int lane = tid & 31;
  const int m    = lane & 15;   // A: row index / B,C: column index
  const int half = lane >> 4;   // lane half selects K/M sub-range
  const int row0 = blockIdx.x * 64 + wid * 16;   // wave's batch-row base

  bf16*     abuf = (bf16*)smem + wid * 16 * 512;                    // 16 KB/wave
  _Float16* zbuf = (_Float16*)(smem + 4 * 16 * 512 * 2) + wid * 16 * 512;
  bf16*     zbbf = (bf16*)(smem + 4 * 16 * 512 * 2) + wid * 16 * 512; // alias

  const float* inrow = input + (size_t)(row0 + m) * INW;
  const float* hrow  = h0 + (size_t)(row0 + m) * OUTW;

  // ================= Controller GRU (hidden 256, in 384) =================
  // Persistent A fragments: con_input = [ci(256) | factor(128)]; con_state.
  v16bf a_ci[12];
#pragma unroll
  for (int kt = 0; kt < 8; ++kt) a_ci[kt] = a_frag_f32(inrow, kt * 32, half);
#pragma unroll
  for (int kt = 0; kt < 4; ++kt)
    a_ci[8 + kt] = a_frag_f32(hrow + 976, kt * 32, half);
  v16bf a_h[8];
#pragma unroll
  for (int kt = 0; kt < 8; ++kt) a_h[kt] = a_frag_f32(hrow + 512, kt * 32, half);

  // Pass 1: z, r gates; stage r*h (A-operand of the n-GEMM) to LDS.
  for (int j = 0; j < 16; ++j) {
    const int n0 = j * 16;
    v8f xz = vzero8(), xr = vzero8(), hz = vzero8(), hr = vzero8();
#pragma unroll
    for (int kt = 0; kt < 12; ++kt) {
      xz = wmma_bf16(a_ci[kt], b_frag(conWih, 384, n0 + m, kt * 32, half), xz);
      xr = wmma_bf16(a_ci[kt], b_frag(conWih, 384, 256 + n0 + m, kt * 32, half), xr);
    }
#pragma unroll
    for (int kt = 0; kt < 8; ++kt) {
      hz = wmma_bf16(a_h[kt], b_frag(conWhh, 256, n0 + m, kt * 32, half), hz);
      hr = wmma_bf16(a_h[kt], b_frag(conWhh, 256, 256 + n0 + m, kt * 32, half), hr);
    }
    const float bz = conBih[n0 + m] + conBhh[n0 + m];
    const float br = conBih[256 + n0 + m] + conBhh[256 + n0 + m];
#pragma unroll
    for (int v = 0; v < 8; ++v) {
      const int r = v + half * 8;
      const float hv = h0[(size_t)(row0 + r) * OUTW + 512 + n0 + m];
      const float z  = 1.0f / (1.0f + __expf(-(xz[v] + hz[v] + bz)));
      const float rr = 1.0f / (1.0f + __expf(-(xr[v] + hr[v] + br)));
      zbuf[r * 512 + n0 + m] = (_Float16)z;
      abuf[r * 512 + n0 + m] = (__bf16)(rr * hv);   // (r ⊙ h) staging
    }
  }

  // Pass 2: n = tanh(x_n + (r⊙h)Wn + b); h' = z*h + (1-z)*n, clip ±5.
  for (int j = 0; j < 16; ++j) {
    const int n0 = j * 16;
    v8f xn = vzero8(), rn = vzero8();
#pragma unroll
    for (int kt = 0; kt < 12; ++kt)
      xn = wmma_bf16(a_ci[kt], b_frag(conWih, 384, 512 + n0 + m, kt * 32, half), xn);
#pragma unroll
    for (int kt = 0; kt < 8; ++kt)
      rn = wmma_bf16(a_frag_lds(abuf + m * 512, kt * 32, half),
                     b_frag(conWhh, 256, 512 + n0 + m, kt * 32, half), rn);
    const float bn = conBih[512 + n0 + m] + conBhh[512 + n0 + m];
#pragma unroll
    for (int v = 0; v < 8; ++v) {
      const int r = v + half * 8;
      const float hv = h0[(size_t)(row0 + r) * OUTW + 512 + n0 + m];
      const float z  = (float)zbuf[r * 512 + n0 + m];
      const float n  = tanhf(xn[v] + rn[v] + bn);
      float hn = z * hv + (1.0f - z) * n;
      hn = fminf(fmaxf(hn, -5.0f), 5.0f);
      out[(size_t)(row0 + r) * OUTW + 512 + n0 + m] = hn;      // con_state'
      zbbf[r * 512 + 256 + n0 + m] = (__bf16)hn;               // A-staging
    }
  }

  // ================= co linear: con_state' (256) -> mean|logvar (64|64) ===
  for (int jp = 0; jp < 4; ++jp) {
    const int n0 = jp * 16;
    v8f am = vzero8(), al = vzero8();
#pragma unroll
    for (int kt = 0; kt < 8; ++kt) {
      v16bf a = a_frag_lds(zbbf + m * 512 + 256, kt * 32, half);
      am = wmma_bf16(a, b_frag(coW, 256, n0 + m, kt * 32, half), am);
      al = wmma_bf16(a, b_frag(coW, 256, 64 + n0 + m, kt * 32, half), al);
    }
    const float bm = coB[n0 + m];
    const float bl = coB[64 + n0 + m];
#pragma unroll
    for (int v = 0; v < 8; ++v) {
      const int r = v + half * 8;
      const float mean = am[v] + bm;
      const float sd   = __expf(0.5f * (al[v] + bl));
      const float ep   = eps[(size_t)(row0 + r) * 64 + n0 + m];
      const float co   = mean + sd * ep;
      out[(size_t)(row0 + r) * OUTW + 768 + n0 + m] = mean;    // co_mean
      out[(size_t)(row0 + r) * OUTW + 832 + n0 + m] = sd;      // co_std
      out[(size_t)(row0 + r) * OUTW + 896 + n0 + m] = co;      // gen_input[0:64]
      abuf[r * 512 + n0 + m] = (__bf16)co;                     // A-staging
    }
  }
  // gen_input tail: ext(16) + zero-pad K 80..95 for the padded weight copy.
#pragma unroll
  for (int v = 0; v < 8; ++v) {
    const int r = v + half * 8;
    const float ex = input[(size_t)(row0 + r) * INW + 256 + m];
    out[(size_t)(row0 + r) * OUTW + 960 + m] = ex;             // gen_input[64:80]
    abuf[r * 512 + 64 + m] = (__bf16)ex;
    abuf[r * 512 + 80 + m] = (__bf16)0.0f;
  }

  // ================= Generator GRU (hidden 512, in 80 padded to 96) =======
  v16bf a_gi[3];
#pragma unroll
  for (int kt = 0; kt < 3; ++kt)
    a_gi[kt] = a_frag_lds(abuf + m * 512, kt * 32, half);
  v16bf a_gh[16];
#pragma unroll
  for (int kt = 0; kt < 16; ++kt) a_gh[kt] = a_frag_f32(hrow, kt * 32, half);

  for (int j = 0; j < 32; ++j) {
    const int n0 = j * 16;
    v8f xz = vzero8(), xr = vzero8(), hz = vzero8(), hr = vzero8();
#pragma unroll
    for (int kt = 0; kt < 3; ++kt) {
      xz = wmma_bf16(a_gi[kt], b_frag(genWih, 96, n0 + m, kt * 32, half), xz);
      xr = wmma_bf16(a_gi[kt], b_frag(genWih, 96, 512 + n0 + m, kt * 32, half), xr);
    }
#pragma unroll
    for (int kt = 0; kt < 16; ++kt) {
      hz = wmma_bf16(a_gh[kt], b_frag(genWhh, 512, n0 + m, kt * 32, half), hz);
      hr = wmma_bf16(a_gh[kt], b_frag(genWhh, 512, 512 + n0 + m, kt * 32, half), hr);
    }
    const float bz = genBih[n0 + m] + genBhh[n0 + m];
    const float br = genBih[512 + n0 + m] + genBhh[512 + n0 + m];
#pragma unroll
    for (int v = 0; v < 8; ++v) {
      const int r = v + half * 8;
      const float hv = h0[(size_t)(row0 + r) * OUTW + n0 + m];
      const float z  = 1.0f / (1.0f + __expf(-(xz[v] + hz[v] + bz)));
      const float rr = 1.0f / (1.0f + __expf(-(xr[v] + hr[v] + br)));
      zbuf[r * 512 + n0 + m] = (_Float16)z;
      abuf[r * 512 + n0 + m] = (__bf16)(rr * hv);
    }
  }

  for (int j = 0; j < 32; ++j) {
    const int n0 = j * 16;
    v8f xn = vzero8(), rn = vzero8();
#pragma unroll
    for (int kt = 0; kt < 3; ++kt)
      xn = wmma_bf16(a_gi[kt], b_frag(genWih, 96, 1024 + n0 + m, kt * 32, half), xn);
#pragma unroll
    for (int kt = 0; kt < 16; ++kt)
      rn = wmma_bf16(a_frag_lds(abuf + m * 512, kt * 32, half),
                     b_frag(genWhh, 512, 1024 + n0 + m, kt * 32, half), rn);
    const float bn = genBih[1024 + n0 + m] + genBhh[1024 + n0 + m];
#pragma unroll
    for (int v = 0; v < 8; ++v) {
      const int r = v + half * 8;
      const float hv = h0[(size_t)(row0 + r) * OUTW + n0 + m];
      const float z  = (float)zbuf[r * 512 + n0 + m];   // read, then slot reused
      const float n  = tanhf(xn[v] + rn[v] + bn);
      float hn = z * hv + (1.0f - z) * n;
      hn = fminf(fmaxf(hn, -5.0f), 5.0f);
      out[(size_t)(row0 + r) * OUTW + n0 + m] = hn;             // gen_state'
      zbbf[r * 512 + n0 + m] = (__bf16)hn;                      // A-staging
    }
  }

  // ================= Factors: gen_state' (512) -> 128 =====================
  for (int j = 0; j < 8; ++j) {
    const int n0 = j * 16;
    v8f acc = vzero8();
#pragma unroll
    for (int kt = 0; kt < 16; ++kt)
      acc = wmma_bf16(a_frag_lds(zbbf + m * 512, kt * 32, half),
                      b_frag(facWn, 512, n0 + m, kt * 32, half), acc);
#pragma unroll
    for (int v = 0; v < 8; ++v) {
      const int r = v + half * 8;
      out[(size_t)(row0 + r) * OUTW + 976 + n0 + m] = acc[v];   // factor
    }
  }
}

// ---------------------------------------------------------------------
extern "C" void kernel_launch(void* const* d_in, const int* in_sizes, int n_in,
                              void* d_out, int out_size, void* d_ws, size_t ws_size,
                              hipStream_t stream) {
  const float* input  = (const float*)d_in[0];
  const float* h0     = (const float*)d_in[1];
  const float* eps    = (const float*)d_in[2];
  const float* genWih = (const float*)d_in[3];
  const float* genBih = (const float*)d_in[4];
  const float* genWhh = (const float*)d_in[5];
  const float* genBhh = (const float*)d_in[6];
  const float* conWih = (const float*)d_in[7];
  const float* conBih = (const float*)d_in[8];
  const float* conWhh = (const float*)d_in[9];
  const float* conBhh = (const float*)d_in[10];
  const float* coW    = (const float*)d_in[11];
  const float* coB    = (const float*)d_in[12];
  const float* facW   = (const float*)d_in[13];
  float* out = (float*)d_out;

  bf16* ws = (bf16*)d_ws;   // needs WS_ELEMS*2 ≈ 3.05 MB of d_ws

  cvt_weights_kernel<<<(WS_ELEMS - 65536 + 255) / 256, 256, 0, stream>>>(
      conWih, conWhh, coW, genWih, genWhh, ws);
  fac_norm_kernel<<<128, 128, 0, stream>>>(facW, ws + OFF_FACN);

  decoder_cell_kernel<<<512, 128, 131072, stream>>>(
      input, h0, eps, genBih, genBhh, conBih, conBhh, coB,
      ws + OFF_CONWIH, ws + OFF_CONWHH, ws + OFF_COW,
      ws + OFF_GENWIH, ws + OFF_GENWHH, ws + OFF_FACN, out);
}
